// HHLTraj_17360257811045
// MI455X (gfx1250) — matrix operand order repaired and checked by reading, not compiled, not run on hardware
//
#include <hip/hip_runtime.h>
#include <hip/hip_bf16.h>
#include <math.h>

// ---------------- problem constants (from reference) ----------------
#define NN   8192   // nodes
#define EE   2048   // hyperedges
#define BB   512    // batch
#define TT   128    // seq len
#define HH   128    // hidden
#define H3   384    // 3*H
#define LL   3      // gru layers
#define BT   (BB*TT)

typedef __attribute__((ext_vector_type(16))) _Float16     v16h;
typedef __attribute__((ext_vector_type(8)))  _Float16     v8h;
typedef __attribute__((ext_vector_type(8)))  float        v8f;
typedef __attribute__((ext_vector_type(4)))  unsigned int v4u;
typedef __attribute__((ext_vector_type(4)))  int          v4i;
typedef __attribute__((ext_vector_type(8)))  int          v8i;

// ---- WMMA fragment loaders (ISA 7.12.2 16-bit layouts, wave32) ----
// A (16xK slice, row-major, ld): lane l<16 holds row M=l, halves
// v[0..7]=K0..7, v[8..15]=K16..23 ; lanes 16-31 hold the K+8 offsets.
__device__ __forceinline__ v16h load_frag_a(const _Float16* p, int ld,
                                            int row0, int k0, int lane) {
  int m  = lane & 15;
  int ko = (lane >> 4) << 3;                 // 0 or 8
  const _Float16* q = p + (size_t)(row0 + m) * ld + k0 + ko;
  v8h lo = *(const v8h*)q;
  v8h hi = *(const v8h*)(q + 16);
  v16h r;
#pragma unroll
  for (int i = 0; i < 8; ++i) { r[i] = lo[i]; r[i + 8] = hi[i]; }
  return r;
}
// B (Kx16 slice) fed from BT[N,K] row-major: lane l<16 holds col N=l,
// v[0..7]=K0..7, v[8..15]=K16..23 ; lanes 16-31 hold the K+8 offsets.
__device__ __forceinline__ v16h load_frag_b(const _Float16* p, int ld,
                                            int col0, int k0, int lane) {
  int n  = lane & 15;
  int ko = (lane >> 4) << 3;                 // 0 or 8
  const _Float16* q = p + (size_t)(col0 + n) * ld + k0 + ko;
  v8h lo = *(const v8h*)q;
  v8h hi = *(const v8h*)(q + 16);
  v16h r;
#pragma unroll
  for (int i = 0; i < 8; ++i) { r[i] = lo[i]; r[i + 8] = hi[i]; }
  return r;
}

// ---------------- generic tiled WMMA GEMM (LDS double-buffered) ----------------
// C[M,Nn] = A[M,K](f16) * BT[Nn,K]^T(f16)  with epilogue:
//   v (+ bias[n]) (* scale[m]) (relu) (+ res[m,n])
// OUTMODE: 0 = f32 [M,Nn], 1 = f16 [M,Nn], 2 = f16 transposed [Nn,M]
// block = 256 threads (8 waves), tile = 128(M) x 64(N), K step 32.
// A/B tiles staged through LDS with coalesced b128 loads, double buffered,
// next-next tile prefetched into L2 (global_prefetch_b8).
template <int OUTMODE, bool BIAS, bool SCALE, bool RELU, bool RES>
__global__ void __launch_bounds__(256)
k_gemm(const _Float16* __restrict__ A, const _Float16* __restrict__ BT_,
       const float* __restrict__ bias, const float* __restrict__ scale,
       const float* __restrict__ res, void* __restrict__ out,
       int M, int Nn, int K) {
  __shared__ _Float16 sA[2][128 * 32];   // 8 KB per buffer
  __shared__ _Float16 sB[2][64 * 32];    // 4 KB per buffer
  const int tid  = threadIdx.x;
  const int lane = tid & 31;
  const int wave = tid >> 5;
  const int row0 = blockIdx.x * 128;
  const int col0 = blockIdx.y * 64;

  auto stage = [&](int buf, int k0) {
    {  // A tile 128x32: 2 threads per row, 16 halves each (2x b128)
      const int r = tid >> 1, c = (tid & 1) << 4;
      const _Float16* src = A + (size_t)(row0 + r) * K + k0 + c;
      v8h x0 = *(const v8h*)src;
      v8h x1 = *(const v8h*)(src + 8);
      *(v8h*)&sA[buf][r * 32 + c]     = x0;
      *(v8h*)&sA[buf][r * 32 + c + 8] = x1;
    }
    {  // B tile 64x32: 4 threads per row, 8 halves each (1x b128)
      const int r = tid >> 2, c = (tid & 3) << 3;
      const _Float16* src = BT_ + (size_t)(col0 + r) * K + k0 + c;
      *(v8h*)&sB[buf][r * 32 + c] = *(const v8h*)src;
    }
  };

  stage(0, 0);
  __syncthreads();

  v8f acc[4] = {};
  const int nk = K >> 5;
  for (int kt = 0; kt < nk; ++kt) {
    const int cur = kt & 1;
    if (kt + 1 < nk) {
      if (kt + 2 < nk)  // warm L2 for the tile after next
        __builtin_prefetch(A + (size_t)(row0 + (tid >> 1)) * K + (kt + 2) * 32, 0, 1);
      stage(cur ^ 1, (kt + 1) * 32);
    }
    v16h a = load_frag_a(sA[cur], 32, wave * 16, 0, lane);
#pragma unroll
    for (int j = 0; j < 4; ++j) {
      v16h b = load_frag_b(sB[cur], 32, j * 16, 0, lane);
      acc[j] = __builtin_amdgcn_wmma_f32_16x16x32_f16(
          false, a, false, b, (short)0, acc[j], false, false);
    }
    __syncthreads();
  }

  // C/D layout: lane<16 -> N=lane, M=base+r ; lane>=16 -> M=base+8+r
  const int nlo = lane & 15;
  const int mb  = row0 + wave * 16 + ((lane >> 4) << 3);
#pragma unroll
  for (int j = 0; j < 4; ++j) {
    const int n = col0 + j * 16 + nlo;
#pragma unroll
    for (int r = 0; r < 8; ++r) {
      float v = acc[j][r];
      const int m = mb + r;
      if (BIAS)  v += bias[n];
      if (SCALE) v *= scale[m];
      if (RELU)  v = fmaxf(v, 0.f);
      if (RES)   v += res[(size_t)m * Nn + n];
      if (OUTMODE == 0)      ((float*)out)[(size_t)m * Nn + n] = v;
      else if (OUTMODE == 1) ((_Float16*)out)[(size_t)m * Nn + n] = (_Float16)v;
      else                   ((_Float16*)out)[(size_t)n * M + m] = (_Float16)v;
    }
  }
}

// ---------------- persistent GRU recurrence ----------------
// One block = 16 batch rows, loops all T steps. Whh^T (as BT[3H,H]) is DMA'd
// into LDS by the Tensor Data Mover (96KB of the 320KB WGP pool); per step:
// gh = h16 @ Whh^T via WMMA (8 waves x 3 N-tiles x K=128), then fused gate
// math from precomputed gx.
__global__ void __launch_bounds__(256)
k_gru_rec(const _Float16* __restrict__ WhhT, const float* __restrict__ bhh,
          const float* __restrict__ gx, _Float16* __restrict__ out16) {
  extern __shared__ char smem[];
  _Float16* sW  = (_Float16*)smem;                         // [384*128] f16 @0
  _Float16* sh  = (_Float16*)(smem + 98304);               // [16*128]  f16
  float*    shf = (float*)   (smem + 98304 + 4096);        // [16*128]  f32
  float*    sgh = (float*)   (smem + 98304 + 4096 + 8192); // [16*384]  f32

  const int tid  = threadIdx.x;
  const int lane = tid & 31;
  const int wave = tid >> 5;
  const int b0   = blockIdx.x * 16;

#if __has_builtin(__builtin_amdgcn_tensor_load_to_lds)
  // TDM: 1-D tile, data_size=2B, 49152 elements, global -> LDS offset 0.
  if (wave == 0) {
    const unsigned long long ga = (unsigned long long)(uintptr_t)WhhT;
    const unsigned nelem = (unsigned)(H3 * HH);           // 49152 f16
    v4u g0;
    g0[0] = 1u;                                           // count=1 (valid, user)
    g0[1] = 0u;                                           // lds_addr = 0 (sW at base)
    g0[2] = (unsigned)(ga & 0xffffffffu);                 // global_addr[31:0] @bit64
    g0[3] = (unsigned)((ga >> 32) & 0x1ffffffu) | (2u << 30);  // addr[56:32], type=2
    v8i g1;
    g1[0] = (int)(1u << 16);                              // data_size=1 (2B), wg_mask=0
    g1[1] = (int)((nelem & 0xffffu) << 16);               // tensor_dim0[15:0] @bit48
    g1[2] = (int)((nelem >> 16) | (1u << 16));            // tensor_dim0[31:16]; tensor_dim1=1
    g1[3] = (int)((nelem & 0xffffu) << 16);               // tile_dim0 @bit112
    g1[4] = 0;                                            // tile_dim1/2 unused
    g1[5] = (int)nelem;                                   // tensor_dim0_stride[31:0]
    g1[6] = 0;
    g1[7] = 0;
    v4i gz = {0, 0, 0, 0};
#if defined(__clang_major__) && __clang_major__ >= 23
    v8i gz8 = {0, 0, 0, 0, 0, 0, 0, 0};
    __builtin_amdgcn_tensor_load_to_lds(g0, g1, gz, gz, gz8, 0);
#else
    __builtin_amdgcn_tensor_load_to_lds(g0, g1, gz, gz, 0);
#endif
    __builtin_amdgcn_s_wait_tensorcnt(0);
  }
#else
  for (int i = tid; i < H3 * HH; i += 256) sW[i] = WhhT[i];
#endif
  for (int i = tid; i < 16 * HH; i += 256) { shf[i] = 0.f; sh[i] = (_Float16)0.f; }
  __syncthreads();

  for (int t = 0; t < TT; ++t) {
#pragma unroll
    for (int j = 0; j < 3; ++j) {
      const int n0 = (wave * 3 + j) * 16;
      v8f acc = {};
#pragma unroll
      for (int k0 = 0; k0 < HH; k0 += 32) {
        v16h a = load_frag_a(sh, HH, 0, k0, lane);
        v16h b = load_frag_b(sW, HH, n0, k0, lane);
        acc = __builtin_amdgcn_wmma_f32_16x16x32_f16(
            false, a, false, b, (short)0, acc, false, false);
      }
      const int n  = n0 + (lane & 15);
      const int mb = (lane >> 4) << 3;
#pragma unroll
      for (int r = 0; r < 8; ++r) sgh[(mb + r) * H3 + n] = acc[r] + bhh[n];
    }
    __syncthreads();

    for (int e = tid; e < 16 * HH; e += 256) {
      const int m = e >> 7, h = e & 127;
      const size_t gi = ((size_t)(b0 + m) * TT + t) * (size_t)H3;
      const float xr = gx[gi + h], xz = gx[gi + HH + h], xn = gx[gi + 2 * HH + h];
      const float hr = sgh[m * H3 + h], hz = sgh[m * H3 + HH + h],
                  hn = sgh[m * H3 + 2 * HH + h];
      const float r  = 1.f / (1.f + expf(-(xr + hr)));
      const float z  = 1.f / (1.f + expf(-(xz + hz)));
      const float nn = tanhf(xn + r * hn);
      const float hnew = (1.f - z) * nn + z * shf[e];
      shf[e] = hnew;
      out16[((size_t)(b0 + m) * TT + t) * HH + h] = (_Float16)hnew;
    }
    __syncthreads();
    for (int e = tid; e < 16 * HH; e += 256) sh[e] = (_Float16)shf[e];
    __syncthreads();
  }
}

// ---------------- small elementwise / data-movement kernels ----------------
__global__ void k_prep_H(const float* __restrict__ Hb, const float* __restrict__ invDE,
                         _Float16* __restrict__ Hs, _Float16* __restrict__ Ht) {
  const long i = (long)blockIdx.x * blockDim.x + threadIdx.x;
  if (i < (long)EE * NN) {
    const int e = (int)(i / NN), n = (int)(i % NN);
    const float v = Hb[i];
    Hs[i] = (_Float16)(v * invDE[e]);        // invDE folded for H @ (.)
    Ht[(size_t)n * EE + e] = (_Float16)v;    // transposed copy for H^T @ (.)
  }
}
__global__ void k_f32_to_f16(const float* __restrict__ in, _Float16* __restrict__ out, int n) {
  const int i = blockIdx.x * blockDim.x + threadIdx.x;
  if (i < n) out[i] = (_Float16)in[i];
}
__global__ void k_transpose_to_f16(const float* __restrict__ in, _Float16* __restrict__ out,
                                   int R, int C) {
  const int i = blockIdx.x * blockDim.x + threadIdx.x;
  if (i < R * C) { const int r = i / C, c = i % C; out[c * R + r] = (_Float16)in[i]; }
}
__global__ void k_gather(const _Float16* __restrict__ x, const int* __restrict__ data,
                         _Float16* __restrict__ emb) {
  const int i = blockIdx.x * blockDim.x + threadIdx.x;
  if (i < BT * HH) {
    const int row = i / HH, h = i & 127;
    emb[i] = x[(size_t)data[row] * HH + h];
  }
}
__global__ void k_residual(const _Float16* __restrict__ emb, const _Float16* __restrict__ out16,
                           const int* __restrict__ len, _Float16* __restrict__ inp) {
  const int i = blockIdx.x * blockDim.x + threadIdx.x;
  if (i < BT * HH) {
    const int t = (i / HH) % TT;
    const int b = i / (HH * TT);
    const float o = (t < len[b]) ? fmaxf((float)out16[i], 0.f) : 0.f;
    inp[i] = (_Float16)((float)emb[i] + o);
  }
}
__global__ void k_final(const _Float16* __restrict__ out16, const int* __restrict__ len,
                        float* __restrict__ out) {
  const int i = blockIdx.x * blockDim.x + threadIdx.x;
  if (i < BB * HH) {
    const int b = i / HH, h = i & 127;
    const int t = len[b] - 1;
    out[i] = tanhf((float)out16[((size_t)b * TT + t) * HH + h]);
  }
}

// ---------------- host orchestration ----------------
extern "C" void kernel_launch(void* const* d_in, const int* in_sizes, int n_in,
                              void* d_out, int out_size, void* d_ws, size_t ws_size,
                              hipStream_t stream) {
  (void)in_sizes; (void)n_in; (void)out_size; (void)ws_size;
  const float* Hbat  = (const float*)d_in[0];
  const float* DV2   = (const float*)d_in[1];
  const float* invDE = (const float*)d_in[2];
  const int*   data  = (const int*)  d_in[3];
  const int*   dlen  = (const int*)  d_in[4];
  const float* poi   = (const float*)d_in[5];
  const float* w1    = (const float*)d_in[6];
  const float* b1    = (const float*)d_in[7];
  const float* w2    = (const float*)d_in[8];
  const float* b2    = (const float*)d_in[9];
  const float* Wih   = (const float*)d_in[10];
  const float* Whh   = (const float*)d_in[11];
  const float* bih   = (const float*)d_in[12];
  const float* bhh   = (const float*)d_in[13];
  float* out = (float*)d_out;

  char* ws = (char*)d_ws; size_t off = 0;
  auto take = [&](size_t bytes) -> char* {
    char* p = ws + off; off += (bytes + 255) & ~(size_t)255; return p;
  };
  _Float16* Ht16   = (_Float16*)take((size_t)NN * EE * 2);   // H^T, f16
  _Float16* Hs16   = (_Float16*)take((size_t)EE * NN * 2);   // invDE*H, f16
  _Float16* poi16  = (_Float16*)take((size_t)NN * HH * 2);
  _Float16* W1T    = (_Float16*)take((size_t)HH * HH * 2);
  _Float16* W2T    = (_Float16*)take((size_t)HH * HH * 2);
  _Float16* Wih16  = (_Float16*)take((size_t)LL * H3 * HH * 2);
  _Float16* Whh16  = (_Float16*)take((size_t)LL * H3 * HH * 2);
  _Float16* Xt16   = (_Float16*)take((size_t)HH * NN * 2);   // [H,N] transposed act
  _Float16* Ut16   = (_Float16*)take((size_t)HH * EE * 2);   // [H,E] transposed act
  _Float16* x1_16  = (_Float16*)take((size_t)NN * HH * 2);
  _Float16* x16    = (_Float16*)take((size_t)NN * HH * 2);
  _Float16* emb16  = (_Float16*)take((size_t)BT * HH * 2);
  _Float16* inp16  = (_Float16*)take((size_t)BT * HH * 2);
  _Float16* out16  = (_Float16*)take((size_t)BT * HH * 2);
  float*    gx     = (float*)   take((size_t)BT * H3 * 4);

  const int TB = 256;
  // --- precision conversion / layout prep ---
  k_prep_H<<<((long)EE * NN + TB - 1) / TB, TB, 0, stream>>>(Hbat, invDE, Hs16, Ht16);
  k_f32_to_f16<<<(NN * HH + TB - 1) / TB, TB, 0, stream>>>(poi, poi16, NN * HH);
  k_f32_to_f16<<<(LL * H3 * HH + TB - 1) / TB, TB, 0, stream>>>(Wih, Wih16, LL * H3 * HH);
  k_f32_to_f16<<<(LL * H3 * HH + TB - 1) / TB, TB, 0, stream>>>(Whh, Whh16, LL * H3 * HH);
  k_transpose_to_f16<<<(HH * HH + TB - 1) / TB, TB, 0, stream>>>(w1, W1T, HH, HH);
  k_transpose_to_f16<<<(HH * HH + TB - 1) / TB, TB, 0, stream>>>(w2, W2T, HH, HH);

  // --- hypergraph conv, G factored: DV2*(H^T*(invDE*(H*(DV2*Y)))) ---
  // A' = DV2 * (poi @ W1 + b1)  -> stored transposed [H,N]
  k_gemm<2, true, true, false, false><<<dim3(NN / 128, HH / 64), 256, 0, stream>>>(
      poi16, W1T, b1, DV2, nullptr, Xt16, NN, HH, HH);
  // U' = (invDE*H) @ A'         -> stored transposed [H,E]
  k_gemm<2, false, false, false, false><<<dim3(EE / 128, HH / 64), 256, 0, stream>>>(
      Hs16, Xt16, nullptr, nullptr, nullptr, Ut16, EE, HH, NN);
  // x1 = relu(DV2 * (H^T @ U')) + poi
  k_gemm<1, false, true, true, true><<<dim3(NN / 128, HH / 64), 256, 0, stream>>>(
      Ht16, Ut16, nullptr, DV2, poi, x1_16, NN, HH, EE);
  // second layer
  k_gemm<2, true, true, false, false><<<dim3(NN / 128, HH / 64), 256, 0, stream>>>(
      x1_16, W2T, b2, DV2, nullptr, Xt16, NN, HH, HH);
  k_gemm<2, false, false, false, false><<<dim3(EE / 128, HH / 64), 256, 0, stream>>>(
      Hs16, Xt16, nullptr, nullptr, nullptr, Ut16, EE, HH, NN);
  k_gemm<1, false, true, false, false><<<dim3(NN / 128, HH / 64), 256, 0, stream>>>(
      Ht16, Ut16, nullptr, DV2, nullptr, x16, NN, HH, EE);

  // --- embedding gather ---
  k_gather<<<(BT * HH + TB - 1) / TB, TB, 0, stream>>>(x16, data, emb16);

  // --- 3-layer GRU with residual-from-embedding ---
  const size_t recLds = 98304 + 4096 + 8192 + 24576;  // 132 KB dynamic LDS
  for (int l = 0; l < LL; ++l) {
    const _Float16* in_l = (l == 0) ? emb16 : inp16;
    // gx = inp @ Wih^T + bih  (M=65536, N=384, K=128)
    k_gemm<0, true, false, false, false><<<dim3(BT / 128, H3 / 64), 256, 0, stream>>>(
        in_l, Wih16 + (size_t)l * H3 * HH, bih + l * H3, nullptr, nullptr,
        gx, BT, H3, HH);
    // sequential recurrence, persistent blocks (32 blocks x 16 batch rows)
    k_gru_rec<<<BB / 16, 256, recLds, stream>>>(
        Whh16 + (size_t)l * H3 * HH, bhh + l * H3, gx, out16);
    if (l < LL - 1)
      k_residual<<<(BT * HH + TB - 1) / TB, TB, 0, stream>>>(emb16, out16, dlen, inp16);
  }

  // --- tanh of last valid timestep ---
  k_final<<<(BB * HH + TB - 1) / TB, TB, 0, stream>>>(out16, dlen, out);
}
